// DSVDD_57234734186774
// MI455X (gfx1250) — matrix operand order repaired and checked by reading, not compiled, not run on hardware
//
#include <hip/hip_runtime.h>

// CDNA5 / gfx1250. wave32. Fused distance GEMM + streaming top-3 softmin.
typedef __attribute__((ext_vector_type(16))) __bf16 v16bf;
typedef __attribute__((ext_vector_type(8)))  float  v8f;

#define BDIM   16
#define NDIM   3136
#define CDIM   112
#define MDIM   3136
#define ROWS   (BDIM * NDIM)        // 50176 output rows (b*N + n)
#define NT     (MDIM / 16)          // 196 center tiles of 16
#define KT     4                    // K padded 112 -> 128 = 4 x 32
#define PACK_ELEMS ((size_t)NT * KT * 32 * 16)   // bf16 elements per packed bank

// ---- prep: per-center squared norms --------------------------------------
__global__ void dsvdd_centnorm(const float* __restrict__ Cb,
                               float* __restrict__ centN) {
  int n = blockIdx.x * blockDim.x + threadIdx.x;
  if (n >= MDIM) return;
  float s = 0.f;
  for (int k = 0; k < CDIM; ++k) {
    float v = Cb[(size_t)k * MDIM + n];
    s = fmaf(v, v, s);
  }
  centN[n] = s;
}

// ---- prep: pack C_bank into WMMA B-fragment order, split bf16 hi/lo ------
// B (32x16, K x N) lane layout chosen here (contiguous k, same as A): lane holds column N = lane&15;
// lanes 0-15 carry K = 0..15, lanes 16-31 carry K = 16..31 (2 bf16 / VGPR).
__global__ void dsvdd_pack(const float* __restrict__ Cb,
                           __bf16* __restrict__ bhi,
                           __bf16* __restrict__ blo) {
  int e = blockIdx.x * blockDim.x + threadIdx.x;   // (nt, t, lane)
  if (e >= NT * KT * 32) return;
  int lane = e & 31;
  int t    = (e >> 5) & 3;
  int nt   = e >> 7;
  int n  = nt * 16 + (lane & 15);
  int kb = t * 32 + ((lane >> 4) << 4);
  size_t off = (size_t)e * 16;
  for (int i = 0; i < 16; ++i) {
    int k = kb + i;
    float v = (k < CDIM) ? Cb[(size_t)k * MDIM + n] : 0.f;
    __bf16 h = (__bf16)v;
    __bf16 l = (__bf16)(v - (float)h);
    bhi[off + i] = h;
    blo[off + i] = l;
  }
}

// ---- sorted ascending top-3 insert (branchless, 5 VALU) ------------------
__device__ __forceinline__ void top3_insert(float d, float& t0, float& t1, float& t2) {
  float m0 = fminf(d, t0);
  float x  = fmaxf(d, t0);
  float m1 = fminf(x, t1);
  float y  = fmaxf(x, t1);
  float m2 = fminf(y, t2);
  t0 = m0; t1 = m1; t2 = m2;
}

// ---- main: 1 wave = 16 rows; loop 196 center tiles; 12 WMMA / tile -------
// Software-pipelined: tile nt's WMMAs overlap with tile nt-1's top-3 update
// (fills the WMMA->VALU hazard slots with useful, co-executable VALU work).
__global__ __launch_bounds__(128) void dsvdd_score(
    const float* __restrict__ phi, const float* __restrict__ centN,
    const __bf16* __restrict__ bhiG, const __bf16* __restrict__ bloG,
    float* __restrict__ out) {
  const int lane    = threadIdx.x & 31;
  const int wave    = threadIdx.x >> 5;
  const int rowBase = (blockIdx.x * 4 + wave) * 16;
  const int r       = lane & 15;
  const int half8   = (lane >> 4) << 3;          // 0 | 8
  const int row     = rowBase + r;

  // Build A fragments (hi/lo split) for all 4 K-tiles; fuse row-norm calc.
  // A (16x32) lane layout chosen here (contiguous k, same as B): row M = lane&15; lanes<16 carry K 0-15,
  // lanes>=16 carry K 16-31.
  v16bf Ah[KT], Al[KT];
  float npart = 0.f;
  #pragma unroll
  for (int t = 0; t < KT; ++t) {
    #pragma unroll
    for (int i = 0; i < 16; ++i) {
      int kk = t * 32 + i + (half8 << 1);
      float v = (kk < CDIM) ? phi[(size_t)row * CDIM + kk] : 0.f;
      npart = fmaf(v, v, npart);
      __bf16 h = (__bf16)v;
      Ah[t][i] = h;
      Al[t][i] = (__bf16)(v - (float)h);
    }
  }
  // lane L and L+16 cover disjoint halves of row L's K range.
  float rowNorm = npart + __shfl_xor(npart, 16, 32);
  // Accumulator slot j holds row (half8 + j): fetch that row's norm.
  float nj[8];
  #pragma unroll
  for (int j = 0; j < 8; ++j) nj[j] = __shfl(rowNorm, half8 + j, 32);

  // Track top-3 of e = cn - 2*dot per slot (nj[j] is slot-invariant, sqrt is
  // monotone -> same selection as top-3 of dist; add nj / sqrt at the end).
  float t0[8], t1[8], t2[8];
  #pragma unroll
  for (int j = 0; j < 8; ++j) { t0[j] = 3.4e38f; t1[j] = 3.4e38f; t2[j] = 3.4e38f; }

  const v16bf* Bh = (const v16bf*)bhiG;
  const v16bf* Bl = (const v16bf*)bloG;

  // --- prologue: tile 0 ---
  v8f accP;
  float cnP;
  {
    v8f acc = {0.f, 0.f, 0.f, 0.f, 0.f, 0.f, 0.f, 0.f};
    #pragma unroll
    for (int t = 0; t < KT; ++t) {
      size_t idx = (size_t)t * 32 + lane;
      v16bf bh = Bh[idx];
      v16bf bl = Bl[idx];
      acc = __builtin_amdgcn_wmma_f32_16x16x32_bf16(false, Ah[t], false, bh,
                                                    (short)0, acc, false, false);
      acc = __builtin_amdgcn_wmma_f32_16x16x32_bf16(false, Ah[t], false, bl,
                                                    (short)0, acc, false, false);
      acc = __builtin_amdgcn_wmma_f32_16x16x32_bf16(false, Al[t], false, bh,
                                                    (short)0, acc, false, false);
    }
    accP = acc;
    cnP  = centN[r];
  }

  // --- steady state: issue tile nt, consume tile nt-1 ---
  #pragma unroll 2
  for (int nt = 1; nt < NT; ++nt) {
    if (nt + 1 < NT) {  // pull next B tile toward the WGP
      __builtin_prefetch(&Bh[((size_t)(nt + 1) * KT) * 32 + lane], 0, 0);
      __builtin_prefetch(&Bl[((size_t)(nt + 1) * KT) * 32 + lane], 0, 0);
    }
    v8f acc = {0.f, 0.f, 0.f, 0.f, 0.f, 0.f, 0.f, 0.f};
    #pragma unroll
    for (int t = 0; t < KT; ++t) {
      size_t idx = ((size_t)nt * KT + t) * 32 + lane;
      v16bf bh = Bh[idx];
      v16bf bl = Bl[idx];
      // 3-pass split-bf16 fp32 emulation: Ah*Bh + Ah*Bl + Al*Bh
      acc = __builtin_amdgcn_wmma_f32_16x16x32_bf16(false, Ah[t], false, bh,
                                                    (short)0, acc, false, false);
      acc = __builtin_amdgcn_wmma_f32_16x16x32_bf16(false, Ah[t], false, bl,
                                                    (short)0, acc, false, false);
      acc = __builtin_amdgcn_wmma_f32_16x16x32_bf16(false, Al[t], false, bh,
                                                    (short)0, acc, false, false);
    }
    float cn = centN[nt * 16 + r];
    // consume previous tile's accumulator (no RAW on fresh WMMA dest)
    #pragma unroll
    for (int j = 0; j < 8; ++j) {
      float e = fmaf(-2.f, accP[j], cnP);
      top3_insert(e, t0[j], t1[j], t2[j]);
    }
    accP = acc;
    cnP  = cn;
  }
  // --- epilogue: consume last tile ---
  #pragma unroll
  for (int j = 0; j < 8; ++j) {
    float e = fmaf(-2.f, accP[j], cnP);
    top3_insert(e, t0[j], t1[j], t2[j]);
  }

  // Butterfly top-3 merge across the 16 lanes sharing each row set
  // (masks 1,2,4,8 keep lanes within their half; operand sets stay disjoint).
  #pragma unroll
  for (int j = 0; j < 8; ++j) {
    #pragma unroll
    for (int mask = 1; mask <= 8; mask <<= 1) {
      float o0 = __shfl_xor(t0[j], mask, 32);
      float o1 = __shfl_xor(t1[j], mask, 32);
      float o2 = __shfl_xor(t2[j], mask, 32);
      top3_insert(o0, t0[j], t1[j], t2[j]);
      top3_insert(o1, t0[j], t1[j], t2[j]);
      top3_insert(o2, t0[j], t1[j], t2[j]);
    }
  }

  if (r == 0) {   // lanes 0 and 16 each write 8 rows
    #pragma unroll
    for (int j = 0; j < 8; ++j) {
      float d0 = sqrtf(fmaxf(nj[j] + t0[j], 0.f));
      float d1 = sqrtf(fmaxf(nj[j] + t1[j], 0.f));
      float d2 = sqrtf(fmaxf(nj[j] + t2[j], 0.f));
      // softmin weight of nearest * nearest distance (numerically stable)
      float s = d0 / (1.f + expf(d0 - d1) + expf(d0 - d2));
      out[rowBase + half8 + j] = s;
    }
  }
}

extern "C" void kernel_launch(void* const* d_in, const int* in_sizes, int n_in,
                              void* d_out, int out_size, void* d_ws, size_t ws_size,
                              hipStream_t stream) {
  const float* phi = (const float*)d_in[0];   // (16, 3136, 112) fp32
  const float* Cb  = (const float*)d_in[1];   // (112, 3136) fp32
  (void)in_sizes; (void)n_in; (void)out_size; (void)ws_size;  // h fixed = 56
  float* out = (float*)d_out;                 // (16,1,56,56) fp32

  // workspace: packed hi bank | packed lo bank | center norms  (~1.62 MB)
  __bf16* bhi   = (__bf16*)d_ws;
  __bf16* blo   = bhi + PACK_ELEMS;
  float*  centN = (float*)(blo + PACK_ELEMS);

  dsvdd_centnorm<<<(MDIM + 255) / 256, 256, 0, stream>>>(Cb, centN);
  dsvdd_pack<<<(NT * KT * 32 + 255) / 256, 256, 0, stream>>>(Cb, bhi, blo);
  dsvdd_score<<<ROWS / 64, 128, 0, stream>>>(phi, centN, bhi, blo, out);
}
